// PointVolSDF_3822520893985
// MI455X (gfx1250) — matrix operand order repaired and checked by reading, not compiled, not run on hardware
//
#include <hip/hip_runtime.h>
#include <math.h>

// ---------------------------------------------------------------------------
// PointVolSDF forward on MI455X (gfx1250, wave32, WMMA).
// Compute-bound (~120 GFLOP vs ~100MB HBM traffic) -> bf16 WMMA GEMM chain.
// v2: 64 rows/block + each wave owns 2 N-tiles x 4 M-tiles so every B
//     fragment is loaded exactly once per block (weight L2 traffic ~4x lower).
// ---------------------------------------------------------------------------

typedef __attribute__((ext_vector_type(16))) __bf16       v16bf;
typedef __attribute__((ext_vector_type(8)))  float        v8f;
typedef __attribute__((ext_vector_type(4)))  unsigned int v4u;

#define P_PTS 32768
#define KNB   8
#define NFEAT 32
#define HID   256
#define SLOPE 0.01f

// LDS activation row strides (bf16 elements). 264*2B = 528B = 33*16B -> keeps
// 16B alignment and rotates banks by 4 dwords/row (no 16-way conflicts).
#define FSTR  264
#define RSTR0 328   // 328*2 = 656 = 41*16
#define RSTR  264

// workspace byte offsets (bf16-swizzled weights, then f32 agg_feat)
#define OFF_W0   0u            // 64*256*2   = 32768
#define OFF_W1   32768u        // 256*256*2  = 131072
#define OFF_W2   163840u
#define OFF_W3   294912u
#define OFF_RW0  425984u       // 320*256*2  = 163840
#define OFF_RW1  589824u       // 256*256*2
#define OFF_AGG  720896u       // 32768*256*4 = 32 MB

union Frag16 { v16bf v; v4u u[2]; };

// A fragment (16x32 bf16, M x K) from LDS per ISA 7.12.2:
// lanes 0-15: M=lane, elems {0..7}=K0..7, {8..15}=K16..23
// lanes16-31: M=lane-16, elems {0..7}=K8..15, {8..15}=K24..31
__device__ __forceinline__ v16bf loadA(const __bf16* act, int stride, int mtile,
                                       int kt, int lane) {
  int m  = (mtile << 4) + (lane & 15);
  int kb = (kt << 5) + ((lane >> 4) << 3);
  const __bf16* base = act + m * stride + kb;
  Frag16 f;
  f.u[0] = *reinterpret_cast<const v4u*>(base);       // K kb..kb+7
  f.u[1] = *reinterpret_cast<const v4u*>(base + 16);  // K kb+16..kb+23
  return f.v;
}

// B fragment (32x16 bf16, K x N), pre-swizzled: frag = 32 lanes x 16 bf16.
__device__ __forceinline__ v16bf loadB(const __bf16* __restrict__ W, int frag,
                                       int lane) {
  const __bf16* base = W + (frag << 9) + (lane << 4);
  Frag16 f;
  f.u[0] = *reinterpret_cast<const v4u*>(base);
  f.u[1] = *reinterpret_cast<const v4u*>(base + 8);
  return f.v;
}

// One 256-wide layer over a 64-row activation tile (4 M-tiles).
// 8 waves x 2 N-tiles = 16 N-tiles; each wave computes its 2 N-tiles for ALL
// 4 M-tiles, so each B fragment is loaded once per block and reused 4x from
// registers; A fragments come from LDS (shared across waves).
__device__ __forceinline__ void layer256(const __bf16* __restrict__ W,
                                         const float* __restrict__ bias,
                                         const __bf16* actIn, int strideIn,
                                         __bf16* actOut, int strideOut,
                                         int KT, bool doLeaky, int wave, int lane) {
  int n0 = wave << 1;  // first owned ntile
  float bv0 = bias[(n0 << 4) + (lane & 15)];
  float bv1 = bias[((n0 + 1) << 4) + (lane & 15)];
  v8f acc[2][4];
#pragma unroll
  for (int mt = 0; mt < 4; ++mt) {
    acc[0][mt] = (v8f){bv0, bv0, bv0, bv0, bv0, bv0, bv0, bv0};
    acc[1][mt] = (v8f){bv1, bv1, bv1, bv1, bv1, bv1, bv1, bv1};
  }
  for (int kt = 0; kt < KT; ++kt) {
    v16bf b0 = loadB(W, (kt << 4) + n0, lane);
    v16bf b1 = loadB(W, (kt << 4) + n0 + 1, lane);
#pragma unroll
    for (int mt = 0; mt < 4; ++mt) {
      v16bf a = loadA(actIn, strideIn, mt, kt, lane);
      acc[0][mt] = __builtin_amdgcn_wmma_f32_16x16x32_bf16(false, a, false, b0,
                                                           (short)0, acc[0][mt],
                                                           false, false);
      acc[1][mt] = __builtin_amdgcn_wmma_f32_16x16x32_bf16(false, a, false, b1,
                                                           (short)0, acc[1][mt],
                                                           false, false);
    }
  }
  // D layout: lane n=l&15, VGPR v -> m = v + (l>=16 ? 8 : 0)
#pragma unroll
  for (int nt = 0; nt < 2; ++nt) {
    int n = ((n0 + nt) << 4) + (lane & 15);
#pragma unroll
    for (int mt = 0; mt < 4; ++mt) {
      int mb = (mt << 4) + ((lane >> 4) << 3);
#pragma unroll
      for (int v = 0; v < 8; ++v) {
        float x = acc[nt][mt][v];
        if (doLeaky) x = (x >= 0.f) ? x : SLOPE * x;
        actOut[(mb + v) * strideOut + n] = (__bf16)x;
      }
    }
  }
}

// ---------------------------------------------------------------------------
// Prep: f32 row-major [Korig x 256] -> bf16 fragment-swizzled [Kpad x 256].
// dst element e: frag f=e>>9, lane=(e>>4)&31, elem=e&15;
//   col = (f&15)*16 + (lane&15); krow = (f>>4)*32 + (lane>=16?8:0) + (elem&7) + (elem>=8?16:0).
// ---------------------------------------------------------------------------
__global__ void swizzle_w_kernel(const float* __restrict__ src,
                                 __bf16* __restrict__ dst, int Korig) {
  int e = blockIdx.x * 256 + threadIdx.x;
  int f = e >> 9;
  int lane = (e >> 4) & 31;
  int elem = e & 15;
  int col  = ((f & 15) << 4) + (lane & 15);
  int krow = ((f >> 4) << 5) + ((lane >= 16) ? 8 : 0) + (elem & 7) + ((elem >> 3) << 4);
  float v = (krow < Korig) ? src[krow * 256 + col] : 0.0f;
  dst[e] = (__bf16)v;
}

// ---------------------------------------------------------------------------
// F-MLP: 4096 blocks x 256 thr; block = 64 rows = 8 points x 8 neighbors.
// ---------------------------------------------------------------------------
__global__ __launch_bounds__(256, 1) void f_mlp_kernel(
    const float* __restrict__ shading_pts, const int* __restrict__ neighbor_idx,
    const int* __restrict__ valid_mask, const float* __restrict__ neural_pts,
    const float* __restrict__ neural_feats,
    const __bf16* __restrict__ W0, const float* __restrict__ b0,
    const __bf16* __restrict__ W1, const float* __restrict__ b1,
    const __bf16* __restrict__ W2, const float* __restrict__ b2,
    const __bf16* __restrict__ W3, const float* __restrict__ b3,
    const float* __restrict__ T_W, const float* __restrict__ T_b,
    float* __restrict__ agg_feat, float* __restrict__ out) {
  __shared__ __bf16 actA[64 * FSTR];
  __shared__ __bf16 actB[64 * FSTR];
  __shared__ float  wgt[64];
  __shared__ float  sdfArr[64];

  int tid  = threadIdx.x;
  int wave = tid >> 5, lane = tid & 31;
  int r0 = blockIdx.x << 6;   // first (p,k) row
  int p0 = r0 >> 3;           // first point (8 per block)

  // Phase 1: build 64x64 embedded input tile (3 pos + 24 sin/cos + 32 feat + 5 pad)
  {
    int row = tid >> 2;   // 0..63
    int j   = tid & 3;    // 4 threads/row, 16 cols each
    int r = r0 + row;
    int p = r >> 3, k = r & 7;
    int idx   = neighbor_idx[p * KNB + k];
    int valid = valid_mask[p * KNB + k];
    float xv[3];
    xv[0] = shading_pts[p * 3 + 0] - neural_pts[idx * 3 + 0];
    xv[1] = shading_pts[p * 3 + 1] - neural_pts[idx * 3 + 1];
    xv[2] = shading_pts[p * 3 + 2] - neural_pts[idx * 3 + 2];
    if (j == 0) {
      float d = sqrtf(xv[0]*xv[0] + xv[1]*xv[1] + xv[2]*xv[2]);
      d = fmaxf(d, 1e-12f);
      wgt[row] = valid ? (1.0f / d) : 0.0f;
    }
#pragma unroll
    for (int c = 0; c < 16; ++c) {
      int col = j * 16 + c;
      float val;
      if (col < 3) {
        val = xv[col];
      } else if (col < 27) {               // [l, {sin,cos}, dim] flattened
        int e = col - 3, l = e / 6, rm = e % 6, trig = rm / 3, d = rm % 3;
        float arg = xv[d] * (float)(1 << l);
        val = trig ? __cosf(arg) : __sinf(arg);
      } else if (col < 59) {
        val = neural_feats[idx * NFEAT + (col - 27)];
      } else {
        val = 0.0f;
      }
      actA[row * FSTR + col] = (__bf16)val;
    }
    // warm weight streams for the GEMM chain (global_prefetch_b8)
    __builtin_prefetch(W1 + tid * 256, 0, 2);
    __builtin_prefetch(W2 + tid * 256, 0, 1);
  }
  __syncthreads();

  layer256(W0, b0, actA, FSTR, actB, FSTR, 2, true,  wave, lane); __syncthreads();
  layer256(W1, b1, actB, FSTR, actA, FSTR, 8, true,  wave, lane); __syncthreads();
  layer256(W2, b2, actA, FSTR, actB, FSTR, 8, true,  wave, lane); __syncthreads();
  layer256(W3, b3, actB, FSTR, actA, FSTR, 8, false, wave, lane); __syncthreads();
  // actA = feat [64][256] (bf16)

  // T head: sdf per (p,k) row
  if (tid < 64) {
    float s = T_b[0];
    for (int c = 0; c < HID; ++c) s += (float)actA[tid * FSTR + c] * T_W[c];
    sdfArr[tid] = s;
  }
  __syncthreads();

  // inverse-distance aggregation over K: feat -> workspace, sdf -> out[:,0]
  for (int rep = 0; rep < 8; ++rep) {
    int item = tid + (rep << 8);
    int pt = item >> 8;      // 0..7
    int c  = item & 255;
    float norm = 0.f, s = 0.f;
#pragma unroll
    for (int k = 0; k < 8; ++k) {
      float w = wgt[(pt << 3) + k];
      norm += w;
      s += w * (float)actA[((pt << 3) + k) * FSTR + c];
    }
    agg_feat[(p0 + pt) * HID + c] = s / norm;
  }
  if (tid < 8) {
    float norm = 0.f, s = 0.f;
#pragma unroll
    for (int k = 0; k < 8; ++k) {
      float w = wgt[(tid << 3) + k];
      norm += w;
      s += w * sdfArr[(tid << 3) + k];
    }
    out[(p0 + tid) * 4] = s / norm;
  }
}

// ---------------------------------------------------------------------------
// R-MLP: 512 blocks x 256 thr; block = 64 shading points.
// ---------------------------------------------------------------------------
__global__ __launch_bounds__(256, 1) void r_mlp_kernel(
    const float* __restrict__ ray_dirs, const float* __restrict__ agg_feat,
    const __bf16* __restrict__ RW0, const float* __restrict__ Rb0,
    const __bf16* __restrict__ RW1, const float* __restrict__ Rb1,
    const float* __restrict__ RW2, const float* __restrict__ Rb2,
    float* __restrict__ out) {
  __shared__ __bf16 actA[64 * RSTR0];
  __shared__ __bf16 actB[64 * RSTR];
  __shared__ __bf16 actC[64 * RSTR];
  int tid  = threadIdx.x;
  int wave = tid >> 5, lane = tid & 31;
  int prow0 = blockIdx.x << 6;

  // Phase 1: 64x320 input (3 dir + 36 sin/cos + 256 agg_feat + 25 pad)
  {
    int row = tid >> 2, j = tid & 3;
    int p = prow0 + row;
    float dv[3] = {ray_dirs[p * 3], ray_dirs[p * 3 + 1], ray_dirs[p * 3 + 2]};
    for (int c = 0; c < 80; ++c) {
      int col = j * 80 + c;
      float val;
      if (col < 3) {
        val = dv[col];
      } else if (col < 39) {
        int e = col - 3, l = e / 6, rm = e % 6, trig = rm / 3, d = rm % 3;
        float arg = dv[d] * (float)(1 << l);
        val = trig ? __cosf(arg) : __sinf(arg);
      } else if (col < 295) {
        val = agg_feat[p * HID + (col - 39)];
      } else {
        val = 0.0f;
      }
      actA[row * RSTR0 + col] = (__bf16)val;
    }
    __builtin_prefetch(RW1 + tid * 256, 0, 2);
  }
  __syncthreads();

  layer256(RW0, Rb0, actA, RSTR0, actB, RSTR, 10, true, wave, lane); __syncthreads();
  layer256(RW1, Rb1, actB, RSTR,  actC, RSTR,  8, true, wave, lane); __syncthreads();

  // 256 -> 3 head + sigmoid
  if (tid < 192) {
    int row = tid / 3, o = tid % 3;
    float s = Rb2[o];
    for (int c = 0; c < HID; ++c) s += (float)actC[row * RSTR + c] * RW2[c * 3 + o];
    out[(prow0 + row) * 4 + 1 + o] = 1.0f / (1.0f + __expf(-s));
  }
}

// ---------------------------------------------------------------------------
extern "C" void kernel_launch(void* const* d_in, const int* in_sizes, int n_in,
                              void* d_out, int out_size, void* d_ws, size_t ws_size,
                              hipStream_t stream) {
  const float* shading_pts  = (const float*)d_in[0];
  const float* ray_dirs     = (const float*)d_in[1];
  const int*   neighbor_idx = (const int*)d_in[2];
  const int*   valid_mask   = (const int*)d_in[3];
  const float* neural_pts   = (const float*)d_in[4];
  const float* neural_feats = (const float*)d_in[5];
  const float* F_W0 = (const float*)d_in[6];  const float* F_b0 = (const float*)d_in[7];
  const float* F_W1 = (const float*)d_in[8];  const float* F_b1 = (const float*)d_in[9];
  const float* F_W2 = (const float*)d_in[10]; const float* F_b2 = (const float*)d_in[11];
  const float* F_W3 = (const float*)d_in[12]; const float* F_b3 = (const float*)d_in[13];
  const float* T_W  = (const float*)d_in[14]; const float* T_b  = (const float*)d_in[15];
  const float* R_W0 = (const float*)d_in[16]; const float* R_b0 = (const float*)d_in[17];
  const float* R_W1 = (const float*)d_in[18]; const float* R_b1 = (const float*)d_in[19];
  const float* R_W2 = (const float*)d_in[20]; const float* R_b2 = (const float*)d_in[21];

  char* ws = (char*)d_ws;
  __bf16* W0s  = (__bf16*)(ws + OFF_W0);
  __bf16* W1s  = (__bf16*)(ws + OFF_W1);
  __bf16* W2s  = (__bf16*)(ws + OFF_W2);
  __bf16* W3s  = (__bf16*)(ws + OFF_W3);
  __bf16* RW0s = (__bf16*)(ws + OFF_RW0);
  __bf16* RW1s = (__bf16*)(ws + OFF_RW1);
  float*  aggF = (float*)(ws + OFF_AGG);
  float*  out  = (float*)d_out;

  // weight swizzle (grid = Kpad blocks of 256)
  swizzle_w_kernel<<<64,  256, 0, stream>>>(F_W0, W0s, 59);
  swizzle_w_kernel<<<256, 256, 0, stream>>>(F_W1, W1s, 256);
  swizzle_w_kernel<<<256, 256, 0, stream>>>(F_W2, W2s, 256);
  swizzle_w_kernel<<<256, 256, 0, stream>>>(F_W3, W3s, 256);
  swizzle_w_kernel<<<320, 256, 0, stream>>>(R_W0, RW0s, 295);
  swizzle_w_kernel<<<256, 256, 0, stream>>>(R_W1, RW1s, 256);

  f_mlp_kernel<<<(P_PTS * KNB) / 64, 256, 0, stream>>>(
      shading_pts, neighbor_idx, valid_mask, neural_pts, neural_feats,
      W0s, F_b0, W1s, F_b1, W2s, F_b2, W3s, F_b3, T_W, T_b, aggF, out);

  r_mlp_kernel<<<P_PTS / 64, 256, 0, stream>>>(
      ray_dirs, aggF, RW0s, R_b0, RW1s, R_b1, R_W2, R_b2, out);
}
